// BiDAF_68513318306105
// MI455X (gfx1250) — matrix operand order repaired and checked by reading, not compiled, not run on hardware
//
#include <hip/hip_runtime.h>
#include <hip/hip_bf16.h>
#include <math.h>

// ---------------------------------------------------------------------------
// BiDAF forward for gfx1250 (MI455X). Big GEMMs on v_wmma_f32_16x16x32_f16
// with pre-converted zero-padded f16 operands; tile staging uses gfx1250
// async global->LDS copies (GLOBAL_LOAD_ASYNC_TO_LDS_B128 + s_wait_asynccnt),
// falling back to vectorized load/ds_store otherwise. Recurrent LSTM scans as
// per-batch persistent blocks, attention fused.
// ---------------------------------------------------------------------------

typedef _Float16 v16h __attribute__((ext_vector_type(16)));
typedef _Float16 v8h  __attribute__((ext_vector_type(8)));
typedef float    v8f  __attribute__((ext_vector_type(8)));
typedef int      i4v  __attribute__((ext_vector_type(4)));

#if defined(__AMDGCN__) && \
    __has_builtin(__builtin_amdgcn_global_load_async_to_lds_b128) && \
    __has_builtin(__builtin_amdgcn_s_wait_asynccnt)
#define USE_ASYNC_LDS 1
// (AS1 int4* src, AS3 int4* dst, imm offset, imm cpol)
#define ASYNC_CP128(gsrc, ldst)                                        \
  __builtin_amdgcn_global_load_async_to_lds_b128(                      \
      (__attribute__((address_space(1))) i4v*)(void*)(gsrc),           \
      (__attribute__((address_space(3))) i4v*)(void*)(ldst), 0, 0)
#else
#define USE_ASYNC_LDS 0
#endif

#define Bn   64
#define LCn  512
#define LQn  64
#define WLn  16
#define CEn  8
#define OCn  100
#define WEn  300
#define Hn   200
#define D2n  400
#define G4n  800

__device__ __forceinline__ float sigf(float x) { return 1.f / (1.f + __expf(-x)); }

// ---------------------------------------------------------------------------
// 1. char embedding -> conv1d(K=5) -> relu -> maxpool over time. One block per
//    word; writes columns [0,100) of the 400-wide concat row.
// ---------------------------------------------------------------------------
__global__ __launch_bounds__(128) void char_conv(
    const int* __restrict__ ids, const float* __restrict__ cemb,
    const float* __restrict__ convw, const float* __restrict__ convb,
    float* __restrict__ out)
{
  __shared__ float se[WLn * CEn];  // 16 chars x 8 dims
  const int word = blockIdx.x;
  const int t = threadIdx.x;
  {
    const int ci = t >> 3, ei = t & 7;
    const int id = ids[(size_t)word * WLn + ci];
    se[t] = cemb[id * CEn + ei];
  }
  __syncthreads();
  if (t < OCn) {
    float best = 0.f;  // relu(y) >= 0, so 0 is the identity for the max
    for (int p = 0; p < WLn - 4; ++p) {   // 12 valid positions
      float acc = convb[t];
      #pragma unroll
      for (int w = 0; w < 5; ++w)
        #pragma unroll
        for (int e = 0; e < CEn; ++e)
          acc = fmaf(se[(p + w) * CEn + e], convw[(w * CEn + e) * OCn + t], acc);
      best = fmaxf(best, fmaxf(acc, 0.f));
    }
    out[(size_t)word * D2n + t] = best;
  }
}

// ---------------------------------------------------------------------------
// 2. word embedding gather into columns [100,400)
// ---------------------------------------------------------------------------
__global__ __launch_bounds__(320) void word_gather(
    const int* __restrict__ wids, const float* __restrict__ wemb,
    float* __restrict__ out)
{
  const int word = blockIdx.x;
  const int t = threadIdx.x;
  if (t < WEn) {
    const int id = wids[word];
    out[(size_t)word * D2n + OCn + t] = wemb[(size_t)id * WEn + t];
  }
}

// ---------------------------------------------------------------------------
// 3a. f32 -> f16 convert with zero-padding of the K dimension to Kp.
//     dst is (rows, Kp) f16, src is (rows, K) f32.
// ---------------------------------------------------------------------------
__global__ __launch_bounds__(256) void cvt_f16_pad(
    const float* __restrict__ src, _Float16* __restrict__ dst,
    int K, int Kp, size_t n)
{
  const size_t i = (size_t)blockIdx.x * 256 + threadIdx.x;
  if (i < n) {
    const size_t r = i / (size_t)Kp;
    const int k = (int)(i - r * (size_t)Kp);
    dst[i] = (k < K) ? (_Float16)src[r * (size_t)K + k] : (_Float16)0.f;
  }
}

// ---------------------------------------------------------------------------
// 3b. WMMA GEMM:  C[m,n] = act( sum_k A[m,k]*W[n,k] + b0[n] + b1[n] )
//     A16: (M,Kp) f16, W16: (N,Kp) f16, Kp % 64 == 0, M % 256 == 0, N % 16 == 0.
//     Tile: BM=256, BN=16, KT=64; 8 waves; each wave owns two 16-row tiles and
//     issues 4 v_wmma_f32_16x16x32_f16 per K-chunk (shared B fragment).
//     Staging via async global->LDS (ASYNCcnt) when available.
// ---------------------------------------------------------------------------
#define BM 256
#define BN 16
#define KT 64

__global__ __launch_bounds__(256) void gemm_f16wmma(
    const _Float16* __restrict__ A16, const _Float16* __restrict__ W16,
    const float* __restrict__ b0, const float* __restrict__ b1,
    float* __restrict__ C, int M, int N, int Kp, int act)
{
  __shared__ _Float16 sA[BM][KT];  // 32 KB
  __shared__ _Float16 sW[BN][KT];  //  2 KB
  const int tid  = threadIdx.x;
  const int wave = tid >> 5;
  const int lane = tid & 31;
  const int half = lane >> 4;        // 0: lanes 0-15, 1: lanes 16-31
  const int ncol = lane & 15;
  const int m0 = blockIdx.y * BM;
  const int n0 = blockIdx.x * BN;

  v8f acc0 = {}, acc1 = {};
  const int mrow0 = wave * 16 + (lane & 15);
  const int mrow1 = mrow0 + 128;

  for (int k0 = 0; k0 < Kp; k0 += KT) {
    // stage A tile: 256x64 halves = 2048 x 16B, 8 transfers per thread
    #pragma unroll
    for (int v = tid; v < (BM * KT) / 8; v += 256) {
      const int r = v >> 3;
      const int c = (v & 7) * 8;
      const _Float16* gp = &A16[(size_t)(m0 + r) * Kp + k0 + c];
#if USE_ASYNC_LDS
      ASYNC_CP128(gp, &sA[r][c]);
#else
      *(v8h*)&sA[r][c] = *(const v8h*)gp;
#endif
    }
    // stage W tile: 16x64 halves = 128 x 16B
    if (tid < (BN * KT) / 8) {
      const int r = tid >> 3;
      const int c = (tid & 7) * 8;
      const _Float16* gp = &W16[(size_t)(n0 + r) * Kp + k0 + c];
#if USE_ASYNC_LDS
      ASYNC_CP128(gp, &sW[r][c]);
#else
      *(v8h*)&sW[r][c] = *(const v8h*)gp;
#endif
    }
    if (k0 + KT < Kp)
      __builtin_prefetch(&A16[(size_t)(m0 + tid) * Kp + k0 + KT], 0, 3);
#if USE_ASYNC_LDS
    __builtin_amdgcn_s_wait_asynccnt(0);
#endif
    __syncthreads();

    #pragma unroll
    for (int kc = 0; kc < KT; kc += 32) {
      // B fragment: lane half*16+n holds K=kc+half*8+e and kc+16+half*8+e, col n
      v16h bfrag;
      {
        const v8h lo = *(const v8h*)&sW[ncol][kc + half * 8];
        const v8h hi = *(const v8h*)&sW[ncol][kc + 16 + half * 8];
        #pragma unroll
        for (int e = 0; e < 8; ++e) { bfrag[e] = lo[e]; bfrag[8 + e] = hi[e]; }
      }
      // A fragments: two aligned 8-half chunks at [kc+half*8], [kc+16+half*8]
      {
        const v8h lo = *(const v8h*)&sA[mrow0][kc + half * 8];
        const v8h hi = *(const v8h*)&sA[mrow0][kc + 16 + half * 8];
        v16h afrag;
        #pragma unroll
        for (int e = 0; e < 8; ++e) { afrag[e] = lo[e]; afrag[8 + e] = hi[e]; }
        acc0 = __builtin_amdgcn_wmma_f32_16x16x32_f16(
            false, afrag, false, bfrag, (short)0, acc0, false, false);
      }
      {
        const v8h lo = *(const v8h*)&sA[mrow1][kc + half * 8];
        const v8h hi = *(const v8h*)&sA[mrow1][kc + 16 + half * 8];
        v16h afrag;
        #pragma unroll
        for (int e = 0; e < 8; ++e) { afrag[e] = lo[e]; afrag[8 + e] = hi[e]; }
        acc1 = __builtin_amdgcn_wmma_f32_16x16x32_f16(
            false, afrag, false, bfrag, (short)0, acc1, false, false);
      }
    }
    __syncthreads();
  }

  // C/D layout: VGPR r -> row r (lanes 0-15) / row r+8 (lanes 16-31), col=lane&15
  const int col = n0 + ncol;
  const float bias = (b0 ? b0[col] : 0.f) + (b1 ? b1[col] : 0.f);
  #pragma unroll
  for (int r = 0; r < 8; ++r) {
    {
      const int row = m0 + wave * 16 + r + 8 * half;
      float v = acc0[r] + bias;
      if (act == 1)      v = fmaxf(v, 0.f);
      else if (act == 2) v = sigf(v);
      C[(size_t)row * N + col] = v;
    }
    {
      const int row = m0 + 128 + wave * 16 + r + 8 * half;
      float v = acc1[r] + bias;
      if (act == 1)      v = fmaxf(v, 0.f);
      else if (act == 2) v = sigf(v);
      C[(size_t)row * N + col] = v;
    }
  }
}

// ---------------------------------------------------------------------------
// 4. highway combine: x = g*h + (1-g)*x
// ---------------------------------------------------------------------------
__global__ __launch_bounds__(256) void highway_combine(
    float* __restrict__ x, const float* __restrict__ h,
    const float* __restrict__ g, size_t n)
{
  const size_t i = (size_t)blockIdx.x * blockDim.x + threadIdx.x;
  if (i < n) {
    const float gv = g[i];
    x[i] = gv * h[i] + (1.f - gv) * x[i];
  }
}

// ---------------------------------------------------------------------------
// 5. LSTM recurrence. One block per batch element; h/c/gates in LDS; scans the
//    whole sequence. xg rows are (b*L + t)*800, out rows (b*L + t)*400+outoff.
// ---------------------------------------------------------------------------
__global__ __launch_bounds__(256) void lstm_rec(
    const float* __restrict__ xg, const float* __restrict__ Whh,  // 800x200
    float* __restrict__ out, int L, int rev, int outoff)
{
  __shared__ float sh[Hn], sc[Hn], sg[G4n];
  const int b = blockIdx.x;
  const int t = threadIdx.x;
  for (int j = t; j < Hn; j += 256) { sh[j] = 0.f; sc[j] = 0.f; }
  __syncthreads();
  for (int step = 0; step < L; ++step) {
    const int tt = rev ? (L - 1 - step) : step;
    const float* xrow = xg + ((size_t)b * L + tt) * G4n;
    for (int gi = t; gi < G4n; gi += 256) {
      float acc = xrow[gi];
      const float* wr = Whh + (size_t)gi * Hn;
      for (int j = 0; j < Hn; ++j) acc = fmaf(sh[j], wr[j], acc);
      sg[gi] = acc;
    }
    __syncthreads();
    for (int j = t; j < Hn; j += 256) {
      const float ig = sg[j], fg = sg[Hn + j], gg = sg[2 * Hn + j], og = sg[3 * Hn + j];
      const float c = sigf(fg) * sc[j] + sigf(ig) * tanhf(gg);
      const float h = sigf(og) * tanhf(c);
      sc[j] = c; sh[j] = h;
      out[((size_t)b * L + tt) * D2n + outoff + j] = h;
    }
    __syncthreads();
  }
}

// ---------------------------------------------------------------------------
// 6. row dot: out[r] (=|+=) dot(X[r,:D], w) + b
// ---------------------------------------------------------------------------
__global__ __launch_bounds__(256) void rowdot(
    const float* __restrict__ X, const float* __restrict__ w,
    const float* __restrict__ bs, float* __restrict__ out, int D, int accumulate)
{
  __shared__ float red[256];
  const size_t r = blockIdx.x;
  const float* x = X + r * D;
  float acc = 0.f;
  for (int j = threadIdx.x; j < D; j += 256) acc = fmaf(x[j], w[j], acc);
  red[threadIdx.x] = acc;
  __syncthreads();
  for (int s = 128; s > 0; s >>= 1) {
    if (threadIdx.x < s) red[threadIdx.x] += red[threadIdx.x + s];
    __syncthreads();
  }
  if (threadIdx.x == 0) {
    float v = red[0] + (bs ? bs[0] : 0.f);
    if (accumulate) out[r] += v; else out[r] = v;
  }
}

// ---------------------------------------------------------------------------
// 7. attention row: for one (b,i) compute s[j], softmax_j, c2q, and write
//    g[:,0:1200]. Also store raw row max for the q2c softmax.
// ---------------------------------------------------------------------------
__global__ __launch_bounds__(128) void att_row(
    const float* __restrict__ encc, const float* __restrict__ encq,
    const float* __restrict__ cw, const float* __restrict__ qw,
    const float* __restrict__ wcq, const float* __restrict__ bcq_p,
    float* __restrict__ g, float* __restrict__ rowmax)
{
  __shared__ float scw[D2n];   // c[b,i,:] * wcq
  __shared__ float cr[D2n];    // c row
  __shared__ float sa[LQn];    // s row -> exp weights
  __shared__ float ssum;
  const size_t row = blockIdx.x;        // = b*LC + i
  const int b = (int)(row / LCn);
  const float* c = encc + row * D2n;
  for (int d = threadIdx.x; d < D2n; d += 128) {
    const float cv = c[d];
    cr[d] = cv;
    scw[d] = cv * wcq[d];
  }
  __syncthreads();
  if (threadIdx.x < LQn) {
    const int j = threadIdx.x;
    const float* qj = encq + ((size_t)b * LQn + j) * D2n;
    float acc = 0.f;
    for (int d = 0; d < D2n; ++d) acc = fmaf(scw[d], qj[d], acc);
    sa[j] = cw[row] + qw[b * LQn + j] + acc + bcq_p[0];
  }
  __syncthreads();
  if (threadIdx.x == 0) {
    float m = sa[0];
    for (int j = 1; j < LQn; ++j) m = fmaxf(m, sa[j]);
    float sum = 0.f;
    for (int j = 0; j < LQn; ++j) { const float e = __expf(sa[j] - m); sa[j] = e; sum += e; }
    ssum = sum;
    rowmax[row] = m;
  }
  __syncthreads();
  const float inv = 1.f / ssum;
  for (int d = threadIdx.x; d < D2n; d += 128) {
    float acc = 0.f;
    for (int j = 0; j < LQn; ++j)
      acc = fmaf(sa[j], encq[((size_t)b * LQn + j) * D2n + d], acc);
    const float c2q = acc * inv;
    const float cv = cr[d];
    g[row * 1600 + d]            = cv;
    g[row * 1600 + D2n + d]      = c2q;
    g[row * 1600 + 2 * D2n + d]  = cv * c2q;
  }
}

// ---------------------------------------------------------------------------
// 8. q2c: per batch b, softmax_i(rowmax[b,:]) then q2c[d]=sum_i w_i*c[b,i,d];
//    broadcast-write g[:,1200:1600] = c * q2c.
// ---------------------------------------------------------------------------
__global__ __launch_bounds__(256) void att_q2c(
    const float* __restrict__ encc, const float* __restrict__ rowmax,
    float* __restrict__ g)
{
  __shared__ float w[LCn];
  __shared__ float q2c[D2n];
  __shared__ float red[256];
  const int b = blockIdx.x;
  float m = -1e30f;
  for (int i = threadIdx.x; i < LCn; i += 256) m = fmaxf(m, rowmax[b * LCn + i]);
  red[threadIdx.x] = m;
  __syncthreads();
  for (int s = 128; s > 0; s >>= 1) {
    if (threadIdx.x < s) red[threadIdx.x] = fmaxf(red[threadIdx.x], red[threadIdx.x + s]);
    __syncthreads();
  }
  const float gm = red[0];
  __syncthreads();
  float sum = 0.f;
  for (int i = threadIdx.x; i < LCn; i += 256) {
    const float e = __expf(rowmax[b * LCn + i] - gm);
    w[i] = e; sum += e;
  }
  red[threadIdx.x] = sum;
  __syncthreads();
  for (int s = 128; s > 0; s >>= 1) {
    if (threadIdx.x < s) red[threadIdx.x] += red[threadIdx.x + s];
    __syncthreads();
  }
  const float inv = 1.f / red[0];
  __syncthreads();
  for (int d = threadIdx.x; d < D2n; d += 256) {
    float acc = 0.f;
    for (int i = 0; i < LCn; ++i)
      acc = fmaf(w[i], encc[((size_t)b * LCn + i) * D2n + d], acc);
    q2c[d] = acc * inv;
  }
  __syncthreads();
  for (int idx = threadIdx.x; idx < LCn * D2n; idx += 256) {
    const int i = idx / D2n, d = idx % D2n;
    const size_t row = (size_t)b * LCn + i;
    g[row * 1600 + 3 * D2n + d] = encc[row * D2n + d] * q2c[d];
  }
}

// ---------------------------------------------------------------------------
// Host orchestration
// ---------------------------------------------------------------------------
static inline int kpad(int k) { return (k + 63) & ~63; }

extern "C" void kernel_launch(void* const* d_in, const int* in_sizes, int n_in,
                              void* d_out, int out_size, void* d_ws, size_t ws_size,
                              hipStream_t stream)
{
  // ---- inputs (setup_inputs dict order, params flattened in dict order) ----
  const int*   cx   = (const int*)d_in[0];
  const int*   cq   = (const int*)d_in[1];
  const int*   xid  = (const int*)d_in[2];
  const int*   qid  = (const int*)d_in[3];
  const float* cemb = (const float*)d_in[4];
  const float* wemb = (const float*)d_in[5];
  const float* convw = (const float*)d_in[6];
  const float* convb = (const float*)d_in[7];
  const float* hw_lw[2] = {(const float*)d_in[8],  (const float*)d_in[12]};
  const float* hw_lb[2] = {(const float*)d_in[9],  (const float*)d_in[13]};
  const float* hw_gw[2] = {(const float*)d_in[10], (const float*)d_in[14]};
  const float* hw_gb[2] = {(const float*)d_in[11], (const float*)d_in[15]};
  auto L4 = [&](int base, int dir, int k) { return (const float*)d_in[base + dir * 4 + k]; };
  const int CTX = 16, MOD1 = 24, MOD2 = 32, OUTL = 40;
  const float* att_wc  = (const float*)d_in[48];
  const float* att_bc  = (const float*)d_in[49];
  const float* att_wq  = (const float*)d_in[50];
  const float* att_bq  = (const float*)d_in[51];
  const float* att_wcq = (const float*)d_in[52];
  const float* att_bcq = (const float*)d_in[53];
  const float* p1g_w = (const float*)d_in[54];
  const float* p1g_b = (const float*)d_in[55];
  const float* p1m_w = (const float*)d_in[56];
  const float* p1m_b = (const float*)d_in[57];
  const float* p2g_w = (const float*)d_in[58];
  const float* p2g_b = (const float*)d_in[59];
  const float* p2m_w = (const float*)d_in[60];
  const float* p2m_b = (const float*)d_in[61];

  const int MC = Bn * LCn;   // 32768 context rows
  const int MQ = Bn * LQn;   // 4096 query rows

  // ---- workspace layout (fp32 bump allocation; all offsets 16B aligned) ----
  float* ws = (float*)d_ws;
  size_t o = 0;
  float* x0c   = ws + o; o += (size_t)MC * D2n;       // concat/highway, context
  float* x0q   = ws + o; o += (size_t)MQ * D2n;       // concat/highway, query
  float* t1    = ws + o; o += (size_t)MC * G4n;       // xg scratch / highway tmps
  float* enc_c = ws + o; o += (size_t)MC * D2n;
  float* enc_q = ws + o; o += (size_t)MQ * D2n;
  float* gbuf  = ws + o; o += (size_t)MC * 1600;
  float* m1    = ws + o; o += (size_t)MC * D2n;       // also reused for m2
  float* mbuf  = ws + o; o += (size_t)MC * D2n;
  float* cwb   = ws + o; o += (size_t)MC;
  float* qwb   = ws + o; o += (size_t)MQ;
  float* rmax  = ws + o; o += (size_t)MC;
  _Float16* a16 = (_Float16*)(ws + o); o += (size_t)MC * 1600 / 2;  // max rows*Kp halves
  _Float16* w16 = (_Float16*)(ws + o); o += (size_t)G4n * 1600 / 2;
  float* tmpH = t1;
  float* tmpG = t1 + (size_t)MC * D2n;

  float* p1 = (float*)d_out;
  float* p2 = p1 + MC;

  auto cvt = [&](const float* src, size_t rows, int K, int Kp, _Float16* dst) {
    const size_t n = rows * (size_t)Kp;
    cvt_f16_pad<<<dim3((unsigned)((n + 255) / 256)), 256, 0, stream>>>(src, dst, K, Kp, n);
  };
  auto gemm = [&](const _Float16* A16, const _Float16* W16, const float* b0,
                  const float* b1, float* C, int M, int N, int Kp, int act) {
    gemm_f16wmma<<<dim3(N / BN, M / BM), 256, 0, stream>>>(A16, W16, b0, b1, C, M, N, Kp, act);
  };

  // ---- 1. embeddings ----
  char_conv<<<MC, 128, 0, stream>>>(cx, cemb, convw, convb, x0c);
  char_conv<<<MQ, 128, 0, stream>>>(cq, cemb, convw, convb, x0q);
  word_gather<<<MC, 320, 0, stream>>>(xid, wemb, x0c);
  word_gather<<<MQ, 320, 0, stream>>>(qid, wemb, x0q);

  // ---- 2. highway (2 layers) on context and query ----
  const int KpD2 = kpad(D2n);   // 448
  for (int l = 0; l < 2; ++l) {
    cvt(x0c, MC, D2n, KpD2, a16);
    cvt(hw_lw[l], D2n, D2n, KpD2, w16);
    gemm(a16, w16, hw_lb[l], nullptr, tmpH, MC, D2n, KpD2, 1);
    cvt(hw_gw[l], D2n, D2n, KpD2, w16);
    gemm(a16, w16, hw_gb[l], nullptr, tmpG, MC, D2n, KpD2, 2);
    highway_combine<<<((size_t)MC * D2n + 255) / 256, 256, 0, stream>>>(
        x0c, tmpH, tmpG, (size_t)MC * D2n);

    cvt(x0q, MQ, D2n, KpD2, a16);
    cvt(hw_lw[l], D2n, D2n, KpD2, w16);
    gemm(a16, w16, hw_lb[l], nullptr, tmpH, MQ, D2n, KpD2, 1);
    cvt(hw_gw[l], D2n, D2n, KpD2, w16);
    gemm(a16, w16, hw_gb[l], nullptr, tmpG, MQ, D2n, KpD2, 2);
    highway_combine<<<((size_t)MQ * D2n + 255) / 256, 256, 0, stream>>>(
        x0q, tmpH, tmpG, (size_t)MQ * D2n);
  }

  // ---- 3. BiLSTM helper: xg = in @ Wih.T + bih + bhh, then scan ----
  auto bilstm = [&](const float* in, float* out, int base, int Lseq, int Kin) {
    const int Mrows = Bn * Lseq;
    const int Kp = kpad(Kin);
    cvt(in, (size_t)Mrows, Kin, Kp, a16);               // shared by both dirs
    for (int dir = 0; dir < 2; ++dir) {
      cvt(L4(base, dir, 0), G4n, Kin, Kp, w16);
      gemm(a16, w16, L4(base, dir, 2), L4(base, dir, 3), t1, Mrows, G4n, Kp, 0);
      lstm_rec<<<Bn, 256, 0, stream>>>(t1, L4(base, dir, 1), out,
                                       Lseq, dir, dir ? Hn : 0);
    }
  };
  bilstm(x0c, enc_c, CTX, LCn, D2n);
  bilstm(x0q, enc_q, CTX, LQn, D2n);

  // ---- 4. attention flow -> g (B*LC x 1600) ----
  rowdot<<<MC, 256, 0, stream>>>(enc_c, att_wc, att_bc, cwb, D2n, 0);
  rowdot<<<MQ, 256, 0, stream>>>(enc_q, att_wq, att_bq, qwb, D2n, 0);
  att_row<<<MC, 128, 0, stream>>>(enc_c, enc_q, cwb, qwb, att_wcq, att_bcq,
                                  gbuf, rmax);
  att_q2c<<<Bn, 256, 0, stream>>>(enc_c, rmax, gbuf);

  // ---- 5. modeling: m = bilstm(bilstm(g, mod1), mod2) ----
  bilstm(gbuf, m1, MOD1, LCn, 1600);
  bilstm(m1, mbuf, MOD2, LCn, D2n);

  // ---- 6. p1 = g@p1g + m@p1m ----
  rowdot<<<MC, 256, 0, stream>>>(gbuf, p1g_w, p1g_b, p1, 1600, 0);
  rowdot<<<MC, 256, 0, stream>>>(mbuf, p1m_w, p1m_b, p1, D2n, 1);

  // ---- 7. m2 = bilstm(m, out);  p2 = g@p2g + m2@p2m ----
  bilstm(mbuf, m1, OUTL, LCn, D2n);  // m1 reused as m2
  rowdot<<<MC, 256, 0, stream>>>(gbuf, p2g_w, p2g_b, p2, 1600, 0);
  rowdot<<<MC, 256, 0, stream>>>(m1, p2m_w, p2m_b, p2, D2n, 1);

  (void)in_sizes; (void)n_in; (void)out_size; (void)ws_size;
}
